// H2OLlamaAttention_streaming_27951647162567
// MI455X (gfx1250) — matrix operand order repaired and checked by reading, not compiled, not run on hardware
//
#include <hip/hip_runtime.h>
#include <hip/hip_bf16.h>

typedef __attribute__((ext_vector_type(16))) _Float16 v16h;
typedef __attribute__((ext_vector_type(8)))  _Float16 v8h;
typedef __attribute__((ext_vector_type(8)))  float    v8f;

#define S_LEN   2048
#define HIDDEN  4096
#define HEADS   32
#define KV_HEADS 8
#define HID     128
#define KV_DIM  1024
#define HH_SIZE 256
#define RECENT  512
#define SELECTN 1536     /* S - RECENT */
#define KEEPN   768      /* HH_SIZE + RECENT */
#define MASK_VAL (-32000.0f)
#define Q_SCALE  (0.08838834764831845f)  /* 1/sqrt(128), folded into q_f16 */

// ---------------------------------------------------------------- helpers ---
__device__ __forceinline__ v8f wmma16(v16h a, v16h b, v8f c) {
  // D = A(16x32 f16) x B(32x16 f16) + C(16x16 f32)
  return __builtin_amdgcn_wmma_f32_16x16x32_f16(false, a, false, b, (short)0, c,
                                                false, false);
}
__device__ __forceinline__ v8f zero8() {
  v8f z;
#pragma unroll
  for (int i = 0; i < 8; ++i) z[i] = 0.0f;
  return z;
}
// A-fragment (16x32, 16-bit): lane row m = lane%16, g = lane/16.
// halves j<8 -> k = 8g+j ; j>=8 -> k = 16+8g+(j-8)  (two 16B loads)
__device__ __forceinline__ v16h load_afrag(const _Float16* __restrict__ p, int g) {
  v8h a = *(const v8h*)(p + 8 * g);
  v8h b = *(const v8h*)(p + 16 + 8 * g);
  v16h r;
#pragma unroll
  for (int j = 0; j < 8; ++j) { r[j] = a[j]; r[8 + j] = b[j]; }
  return r;
}
// B-fragment (32x16, 16-bit): lane col n = lane%16, g = lane/16,
// halves j<8 -> k = 8g+j ; j>=8 -> k = 16+8g+(j-8)  (two 16B loads)
__device__ __forceinline__ v16h load_bfrag(const _Float16* __restrict__ p, int g) {
  v8h a = *(const v8h*)(p + 8 * g);
  v8h b = *(const v8h*)(p + 16 + 8 * g);
  v16h r;
#pragma unroll
  for (int j = 0; j < 8; ++j) { r[j] = a[j]; r[8 + j] = b[j]; }
  return r;
}

// ------------------------------------------------------------ cvt kernels ---
__global__ __launch_bounds__(256) void cvt_f32_to_f16(
    const float* __restrict__ in, _Float16* __restrict__ out, int n) {
  int i = blockIdx.x * 256 + threadIdx.x;
  if (i < n) out[i] = (_Float16)in[i];
}

// V (S x 1024 f32) -> Vt f16 [1024][S]  (rows = kvh*128+d, contiguous keys)
__global__ __launch_bounds__(256) void build_vt(const float* __restrict__ v,
                                                _Float16* __restrict__ vt) {
  int i = blockIdx.x * 256 + threadIdx.x;
  if (i >= S_LEN * KV_DIM) return;
  int s = i >> 10, c = i & 1023;
  vt[(size_t)c * S_LEN + s] = (_Float16)v[i];
}

// ---------------------------------------------------------------- RoPE ------
__global__ __launch_bounds__(256) void rope_kernel(
    const float* __restrict__ in, _Float16* __restrict__ out_h,
    float* __restrict__ out_f, int nheads, float scale) {
  int i = blockIdx.x * 256 + threadIdx.x;
  int total = S_LEN * nheads * 64;
  if (i >= total) return;
  int d = i & 63;
  int t = i >> 6;
  int h = t % nheads;
  int s = t / nheads;
  size_t b = (size_t)s * (nheads * HID) + (size_t)h * HID;
  float x0 = in[b + d], x1 = in[b + d + 64];
  float invf = __powf(10000.0f, -(float)d * (1.0f / 64.0f));
  float ang = (float)s * invf;
  float cs = __cosf(ang), sn = __sinf(ang);
  float r0 = x0 * cs - x1 * sn;
  float r1 = x1 * cs + x0 * sn;
  out_h[b + d] = (_Float16)(r0 * scale);
  out_h[b + d + 64] = (_Float16)(r1 * scale);
  if (out_f) { out_f[b + d] = r0; out_f[b + d + 64] = r1; }
}

// ------------------------------------------------ generic WMMA GEMM ---------
// C[M x N] (f32) = A[M x K] (f16, row major) x W[N x K]^T (f16, row major)
// one wave computes a 64x64 tile (4x4 grid of 16x16 WMMA tiles).
// launch_bounds(...,1): accumulators stay register-resident (no spills).
// Addressing: keep base + kk expressions so clang proves global addrspace
// and emits global_load_b128 (flat loads would also tick DScnt).
__global__ __launch_bounds__(256, 1) void gemm_wmma(
    const _Float16* __restrict__ A, const _Float16* __restrict__ W,
    float* __restrict__ C, int M, int N, int K) {
  int wave = blockIdx.x * (blockDim.x >> 5) + (threadIdx.x >> 5);
  int lane = threadIdx.x & 31;
  int n0 = lane & 15, g = lane >> 4;
  int tilesN = N >> 6;
  int tm = wave / tilesN, tn = wave % tilesN;
  if (tm * 64 >= M) return;

  v8f acc[4][4];
#pragma unroll
  for (int i = 0; i < 4; ++i)
#pragma unroll
    for (int j = 0; j < 4; ++j) acc[i][j] = zero8();

  const _Float16* abase = A + (size_t)(tm * 64 + n0) * K;
  const _Float16* wbase = W + (size_t)(tn * 64 + n0) * K;
  const size_t rstep = (size_t)16 * K;

  for (int kk = 0; kk < K; kk += 32) {
    v16h af[4], bf[4];
#pragma unroll
    for (int i = 0; i < 4; ++i) af[i] = load_afrag(abase + i * rstep + kk, g);
#pragma unroll
    for (int j = 0; j < 4; ++j) bf[j] = load_bfrag(wbase + j * rstep + kk, g);
#pragma unroll
    for (int i = 0; i < 4; ++i)
#pragma unroll
      for (int j = 0; j < 4; ++j) acc[i][j] = wmma16(af[i], bf[j], acc[i][j]);
  }
  // D layout: lane col n = n0, rows m = r + 8g
#pragma unroll
  for (int i = 0; i < 4; ++i)
#pragma unroll
    for (int j = 0; j < 4; ++j) {
#pragma unroll
      for (int r = 0; r < 8; ++r) {
        size_t row = (size_t)(tm * 64 + i * 16 + r + 8 * g);
        C[row * N + tn * 64 + j * 16 + n0] = acc[i][j][r];
      }
    }
}

// ----------------------------------- attention pass A (flash, S^T tiles) ----
// grid (32 q-tiles of 64, 32 heads), block 128 (4 waves, 16 queries each)
__global__ __launch_bounds__(128, 1) void attn_fwd(
    const _Float16* __restrict__ qf,   // [S][4096] scaled by 1/sqrt(128)
    const _Float16* __restrict__ kf,   // [S][1024]
    const _Float16* __restrict__ vt,   // [1024][S]
    _Float16* __restrict__ attn,       // [S][4096]
    float* __restrict__ mws, float* __restrict__ lws) {
  int h = blockIdx.y;
  int w = threadIdx.x >> 5;
  int lane = threadIdx.x & 31;
  int n0 = lane & 15, g = lane >> 4;
  int qbase = blockIdx.x * 64 + w * 16;
  int kvh = h >> 2;
  int qglob = qbase + n0;

  // preload Q^T B-fragments: lane col = query, contiguous over d
  const _Float16* qrow = qf + (size_t)qglob * HIDDEN + h * HID;
  v16h qb[4];
#pragma unroll
  for (int c = 0; c < 4; ++c) qb[c] = load_bfrag(qrow + c * 32, g);

  v8f o[8];
#pragma unroll
  for (int t = 0; t < 8; ++t) o[t] = zero8();
  float mi = -1e30f, li = 0.0f;

  const _Float16* kbase = kf + (size_t)n0 * KV_DIM + kvh * HID;
  const _Float16* vbase = vt + (size_t)(kvh * HID + n0) * S_LEN;

  int kend = qbase + 16;
  for (int kc = 0; kc < kend; kc += 32) {
    const _Float16* krow0 = kbase + (size_t)kc * KV_DIM;
    const _Float16* krow1 = kbase + (size_t)(kc + 16) * KV_DIM;
    v8f t0 = zero8(), t1 = zero8();
#pragma unroll
    for (int c = 0; c < 4; ++c) {
      v16h ka0 = load_afrag(krow0 + c * 32, g);
      v16h ka1 = load_afrag(krow1 + c * 32, g);
      t0 = wmma16(ka0, qb[c], t0);   // S^T tile: rows=keys, cols=queries
      t1 = wmma16(ka1, qb[c], t1);
    }
    // causal mask + chunk max (column q = n0; row key = kc + 16t + r + 8g)
    float cmax = -1e30f;
#pragma unroll
    for (int r = 0; r < 8; ++r) {
      int key0 = kc + r + 8 * g;
      if (key0 > qglob) t0[r] = MASK_VAL;
      if (key0 + 16 > qglob) t1[r] = MASK_VAL;
      cmax = fmaxf(cmax, fmaxf(t0[r], t1[r]));
    }
    cmax = fmaxf(cmax, __shfl_xor(cmax, 16, 32));
    float mnew = fmaxf(mi, cmax);
    float sc = __expf(mi - mnew);
    float psum = 0.0f;
    v16h pfrag;  // S^T D-regs map 1:1 onto A-fragment halves for P x V
#pragma unroll
    for (int r = 0; r < 8; ++r) {
      float p0 = __expf(t0[r] - mnew);
      float p1 = __expf(t1[r] - mnew);
      psum += p0 + p1;
      pfrag[r] = (_Float16)p0;
      pfrag[8 + r] = (_Float16)p1;
    }
    psum += __shfl_xor(psum, 16, 32);
    li = li * sc + psum;
    mi = mnew;
    // rescale O: O-tile reg r holds query q = r + 8g; stat lives in lane q
    float f[8];
#pragma unroll
    for (int r = 0; r < 8; ++r) f[r] = __shfl(sc, r + 8 * g, 32);
#pragma unroll
    for (int dt = 0; dt < 8; ++dt)
#pragma unroll
      for (int r = 0; r < 8; ++r) o[dt][r] *= f[r];
    // P x V
#pragma unroll
    for (int dt = 0; dt < 8; ++dt) {
      v16h vfrag = load_bfrag(vbase + (size_t)dt * 16 * S_LEN + kc, g);
      o[dt] = wmma16(pfrag, vfrag, o[dt]);
    }
  }
  if (lane < 16) {
    mws[h * S_LEN + qbase + n0] = mi;
    lws[h * S_LEN + qbase + n0] = li;
  }
  float invl = 1.0f / li;
  float f[8];
#pragma unroll
  for (int r = 0; r < 8; ++r) f[r] = __shfl(invl, r + 8 * g, 32);
#pragma unroll
  for (int dt = 0; dt < 8; ++dt)
#pragma unroll
    for (int r = 0; r < 8; ++r) {
      size_t row = (size_t)(qbase + r + 8 * g);
      attn[row * HIDDEN + h * HID + dt * 16 + n0] = (_Float16)(o[dt][r] * f[r]);
    }
}

// ------------------------- attention pass B: hh_score column sums -----------
// grid (32 key-chunks of 64, 32 heads), block 128 (4 waves, 16 keys each)
__global__ __launch_bounds__(128, 1) void attn_hh(
    const _Float16* __restrict__ qf, const _Float16* __restrict__ kf,
    const float* __restrict__ mws, const float* __restrict__ lws,
    float* __restrict__ hh) {
  int h = blockIdx.y;
  int w = threadIdx.x >> 5;
  int lane = threadIdx.x & 31;
  int n0 = lane & 15, g = lane >> 4;
  int kbase = blockIdx.x * 64 + w * 16;
  int kvh = h >> 2;
  const _Float16* krow = kf + (size_t)(kbase + n0) * KV_DIM + kvh * HID;

  // preload the 4 K A-fragments for this wave's 16 keys (reused all q-tiles)
  v16h ka[4];
#pragma unroll
  for (int c = 0; c < 4; ++c) ka[c] = load_afrag(krow + c * 32, g);

  float hsum[8];
#pragma unroll
  for (int r = 0; r < 8; ++r) hsum[r] = 0.0f;

  const _Float16* qbase0 = qf + (size_t)n0 * HIDDEN + h * HID;

  for (int qt = kbase; qt < S_LEN; qt += 16) {
    int q = qt + n0;
    float mq = mws[h * S_LEN + q];
    float invl = 1.0f / lws[h * S_LEN + q];
    const _Float16* qrow = qbase0 + (size_t)qt * HIDDEN;
    v8f t = zero8();
#pragma unroll
    for (int c = 0; c < 4; ++c) {
      v16h qb = load_bfrag(qrow + c * 32, g);
      t = wmma16(ka[c], qb, t);
    }
#pragma unroll
    for (int r = 0; r < 8; ++r) {
      int key = kbase + r + 8 * g;
      float p = (key > q) ? 0.0f : __expf(t[r] - mq) * invl;
      hsum[r] += p;
    }
  }
#pragma unroll
  for (int r = 0; r < 8; ++r) {
    hsum[r] += __shfl_xor(hsum[r], 1, 32);
    hsum[r] += __shfl_xor(hsum[r], 2, 32);
    hsum[r] += __shfl_xor(hsum[r], 4, 32);
    hsum[r] += __shfl_xor(hsum[r], 8, 32);
  }
  if (n0 == 0) {
#pragma unroll
    for (int r = 0; r < 8; ++r) hh[h * S_LEN + kbase + r + 8 * g] = hsum[r];
  }
}

// ------------------------------- top-k (exact ranks, index-sorted) ----------
__global__ __launch_bounds__(256) void topk_kernel(const float* __restrict__ hh,
                                                   int* __restrict__ keep) {
  __shared__ float vals[SELECTN];
  __shared__ unsigned mask[SELECTN / 32];
  int h = blockIdx.x, tid = threadIdx.x;
  for (int i = tid; i < SELECTN; i += 256) vals[i] = hh[h * S_LEN + i];
  for (int i = tid; i < SELECTN / 32; i += 256) mask[i] = 0u;
  __syncthreads();
  for (int i = tid; i < SELECTN; i += 256) {
    float v = vals[i];
    int cnt = 0;
    for (int j = 0; j < SELECTN; ++j) {
      float u = vals[j];
      cnt += (u > v) || (u == v && j < i);   // lax.top_k tie-break: lower index
    }
    if (cnt < HH_SIZE) atomicOr(&mask[i >> 5], 1u << (i & 31));
  }
  __syncthreads();
  for (int i = tid; i < SELECTN; i += 256) {
    if ((mask[i >> 5] >> (i & 31)) & 1u) {
      int pos = 0;
      for (int wd = 0; wd < (i >> 5); ++wd) pos += __popc(mask[wd]);
      pos += __popc(mask[i >> 5] & ((1u << (i & 31)) - 1u));
      keep[h * KEEPN + pos] = i;             // ascending index == sorted topk
    }
  }
  for (int i = tid; i < RECENT; i += 256)
    keep[h * KEEPN + HH_SIZE + i] = SELECTN + i;
}

// ---------------------------------------- gather k_hh / v_hh ----------------
__global__ __launch_bounds__(128) void gather_kv(
    const int* __restrict__ keep, const float* __restrict__ krope,
    const float* __restrict__ v, float* __restrict__ khh,
    float* __restrict__ vhh) {
  int h = blockIdx.y, j = blockIdx.x, d = threadIdx.x;  // 128 threads
  int idx = keep[h * KEEPN + j];
  int kvh = h >> 2;
  size_t dst = ((size_t)(h * KEEPN + j)) * HID + d;
  size_t src = (size_t)idx * KV_DIM + kvh * HID + d;
  khh[dst] = krope[src];
  vhh[dst] = v[src];
}

// ---------------------------------------------------------------------------
extern "C" void kernel_launch(void* const* d_in, const int* in_sizes, int n_in,
                              void* d_out, int out_size, void* d_ws,
                              size_t ws_size, hipStream_t stream) {
  const float* h_in = (const float*)d_in[0];
  const float* Wq = (const float*)d_in[1];
  const float* Wk = (const float*)d_in[2];
  const float* Wv = (const float*)d_in[3];
  const float* Wo = (const float*)d_in[4];
  float* out = (float*)d_out;

  char* ws = (char*)d_ws;
  // workspace layout (bytes, 256-aligned)
  const size_t OFF_H16 = 0;                 // 16,777,216
  const size_t OFF_WQ16 = 16777216;         // 33,554,432
  const size_t OFF_WK16 = 50331648;         // 8,388,608
  const size_t OFF_WV16 = 58720256;         // 8,388,608
  const size_t OFF_WO16 = 67108864;         // 33,554,432
  const size_t OFF_QF32 = 100663296;        // 33,554,432 (reused as ATT16)
  const size_t OFF_ATT16 = OFF_QF32;        // 16,777,216 (q_f32 dead by then)
  const size_t OFF_KF32 = 134217728;        // 8,388,608 (roped in place)
  const size_t OFF_VF32 = 142606336;        // 8,388,608
  const size_t OFF_Q16 = 150994944;         // 16,777,216
  const size_t OFF_K16 = 167772160;         // 4,194,304
  const size_t OFF_VT16 = 171966464;        // 4,194,304
  const size_t OFF_MWS = 176160768;         // 262,144
  const size_t OFF_LWS = 176422912;         // 262,144
  const size_t OFF_KEEP = 176685056;        // 98,304

  _Float16* H16 = (_Float16*)(ws + OFF_H16);
  _Float16* WQ16 = (_Float16*)(ws + OFF_WQ16);
  _Float16* WK16 = (_Float16*)(ws + OFF_WK16);
  _Float16* WV16 = (_Float16*)(ws + OFF_WV16);
  _Float16* WO16 = (_Float16*)(ws + OFF_WO16);
  float* QF32 = (float*)(ws + OFF_QF32);
  _Float16* ATT16 = (_Float16*)(ws + OFF_ATT16);
  float* KF32 = (float*)(ws + OFF_KF32);
  float* VF32 = (float*)(ws + OFF_VF32);
  _Float16* Q16 = (_Float16*)(ws + OFF_Q16);
  _Float16* K16 = (_Float16*)(ws + OFF_K16);
  _Float16* VT16 = (_Float16*)(ws + OFF_VT16);
  float* MWS = (float*)(ws + OFF_MWS);
  float* LWS = (float*)(ws + OFF_LWS);
  int* KEEP = (int*)(ws + OFF_KEEP);

  // d_out sub-regions (float offsets)
  float* hh_out = out + 8388608;            // 32 x 2048
  float* khh_out = out + 8454144;           // 1 x 32 x 768 x 128
  float* vhh_out = out + 11599872;

  // 1) f32 -> f16 conversions
  cvt_f32_to_f16<<<(S_LEN * HIDDEN) / 256, 256, 0, stream>>>(h_in, H16,
                                                             S_LEN * HIDDEN);
  cvt_f32_to_f16<<<(HIDDEN * HIDDEN) / 256, 256, 0, stream>>>(Wq, WQ16,
                                                              HIDDEN * HIDDEN);
  cvt_f32_to_f16<<<(KV_DIM * HIDDEN) / 256, 256, 0, stream>>>(Wk, WK16,
                                                              KV_DIM * HIDDEN);
  cvt_f32_to_f16<<<(KV_DIM * HIDDEN) / 256, 256, 0, stream>>>(Wv, WV16,
                                                              KV_DIM * HIDDEN);
  cvt_f32_to_f16<<<(HIDDEN * HIDDEN) / 256, 256, 0, stream>>>(Wo, WO16,
                                                              HIDDEN * HIDDEN);

  // 2) QKV projections (WMMA)
  gemm_wmma<<<(32 * 64) / 8, 256, 0, stream>>>(H16, WQ16, QF32, S_LEN, HIDDEN,
                                               HIDDEN);
  gemm_wmma<<<(32 * 16) / 8, 256, 0, stream>>>(H16, WK16, KF32, S_LEN, KV_DIM,
                                               HIDDEN);
  gemm_wmma<<<(32 * 16) / 8, 256, 0, stream>>>(H16, WV16, VF32, S_LEN, KV_DIM,
                                               HIDDEN);

  // 3) RoPE (q scaled by 1/sqrt(128); k kept unscaled, roped f32 in place)
  rope_kernel<<<(S_LEN * HEADS * 64) / 256, 256, 0, stream>>>(QF32, Q16,
                                                              nullptr, HEADS,
                                                              Q_SCALE);
  rope_kernel<<<(S_LEN * KV_HEADS * 64) / 256, 256, 0, stream>>>(KF32, K16,
                                                                 KF32, KV_HEADS,
                                                                 1.0f);
  // 4) V^T in f16
  build_vt<<<(S_LEN * KV_DIM) / 256, 256, 0, stream>>>(VF32, VT16);

  // 5) flash attention (WMMA), saves m/l; writes attn f16 (overlaps dead QF32)
  attn_fwd<<<dim3(32, 32), 128, 0, stream>>>(Q16, K16, VT16, ATT16, MWS, LWS);

  // 6) output projection (WMMA) -> d_out
  gemm_wmma<<<(32 * 64) / 8, 256, 0, stream>>>(ATT16, WO16, out, S_LEN, HIDDEN,
                                               HIDDEN);

  // 7) hh_score column sums (WMMA recompute with final m/l)
  attn_hh<<<dim3(32, 32), 128, 0, stream>>>(Q16, K16, MWS, LWS, hh_out);

  // 8) top-k + recent window indices
  topk_kernel<<<HEADS, 256, 0, stream>>>(hh_out, KEEP);

  // 9) gather k_hh / v_hh (f32 roped K, f32 V)
  gather_kv<<<dim3(KEEPN, HEADS), HID, 0, stream>>>(KEEP, KF32, VF32, khh_out,
                                                    vhh_out);
}
